// SelfAttentionHead_78348793414194
// MI455X (gfx1250) — matrix operand order repaired and checked by reading, not compiled, not run on hardware
//
#include <hip/hip_runtime.h>
#include <hip/hip_bf16.h>

typedef __attribute__((ext_vector_type(16))) _Float16 v16h;
typedef __attribute__((ext_vector_type(8)))  _Float16 v8h;
typedef __attribute__((ext_vector_type(8)))  float    v8f;

#define B_  256
#define T_  256
#define C_  384
#define H_  64
#define XS_PITCH 392   // 384 + 8 halves padding -> 16B lane reads hit distinct banks

union AFrag { v16h v; v8h h[2]; };

// ---------------------------------------------------------------------------
// Phase 0: convert + transpose weights fp32 [C,H] -> f16 [H,C]
// so phase-1 B-fragments are contiguous 32B per-lane loads.
// ---------------------------------------------------------------------------
__global__ void wt_convert_kernel(const float* __restrict__ Wq,
                                  const float* __restrict__ Wk,
                                  const float* __restrict__ Wv,
                                  _Float16* __restrict__ Wt) {
  const int per = C_ * H_;  // 24576
  int tid = blockIdx.x * blockDim.x + threadIdx.x;
  if (tid >= 3 * per) return;
  int m = tid / per, rem = tid - m * per;
  int c = rem / H_, h = rem - c * H_;
  const float* W = (m == 0) ? Wq : (m == 1) ? Wk : Wv;
  Wt[(size_t)m * per + (size_t)h * C_ + c] = (_Float16)W[(size_t)c * H_ + h];
}

// ---------------------------------------------------------------------------
// Phase 1: QKV projection. One block = one (batch, 16-row) tile of x.
// x tile staged in LDS as f16; 4 waves each own one 16-col h-tile and run
// 12 unrolled k-chunks x 3 matrices = 36 WMMAs.
// Q,K stored [B,T,H] f16; V stored transposed [B,H,T] f16.
// ---------------------------------------------------------------------------
__global__ __launch_bounds__(128)
void qkv_proj_kernel(const float* __restrict__ x,
                     const _Float16* __restrict__ Wt,   // [3][H][C] f16
                     _Float16* __restrict__ Qg,         // [B][T][H]
                     _Float16* __restrict__ Kg,         // [B][T][H]
                     _Float16* __restrict__ Vt)         // [B][H][T]
{
  __shared__ _Float16 xs[16 * XS_PITCH];

  const int tid = threadIdx.x;
  const int b   = blockIdx.x >> 4;
  const int t0  = (blockIdx.x & 15) << 4;

  // Cooperative stage: x fp32 -> f16 LDS (touch x exactly once)
  const size_t xbase = ((size_t)b * T_ + t0) * C_;
  for (int i = tid; i < 16 * C_; i += 128) {
    int row = i / C_;
    int c   = i - row * C_;
    xs[row * XS_PITCH + c] = (_Float16)x[xbase + (size_t)row * C_ + c];
  }
  __syncthreads();

  const int lane = tid & 31;
  const int g    = lane >> 4;   // half-wave select
  const int ln   = lane & 15;
  const int w    = tid >> 5;    // wave id -> h-tile
  const int n    = (w << 4) + ln;

  const _Float16* wtq = Wt + 0 * (size_t)(H_ * C_) + (size_t)n * C_;
  const _Float16* wtk = Wt + 1 * (size_t)(H_ * C_) + (size_t)n * C_;
  const _Float16* wtv = Wt + 2 * (size_t)(H_ * C_) + (size_t)n * C_;

  v8f cq = {}, ck = {}, cv = {};

#pragma unroll
  for (int kc = 0; kc < C_ / 32; ++kc) {
    // A-fragment of x: lane=row, K in [16g,16g+16) contiguous
    AFrag a;
    const _Float16* xrow = &xs[ln * XS_PITCH + kc * 32 + g * 16];
    a.h[0] = *(const v8h*)(xrow);
    a.h[1] = *(const v8h*)(xrow + 8);
    // B-fragments: lane=col n, K in [16g,16g+16) contiguous
    const int ko = kc * 32 + g * 16;
    v16h bq = *(const v16h*)(wtq + ko);
    v16h bk = *(const v16h*)(wtk + ko);
    v16h bv = *(const v16h*)(wtv + ko);
    cq = __builtin_amdgcn_wmma_f32_16x16x32_f16(false, a.v, false, bq, (short)0, cq, false, false);
    ck = __builtin_amdgcn_wmma_f32_16x16x32_f16(false, a.v, false, bk, (short)0, ck, false, false);
    cv = __builtin_amdgcn_wmma_f32_16x16x32_f16(false, a.v, false, bv, (short)0, cv, false, false);
  }

  // C-layout: lane holds col n, rows t = r + 8g
  const size_t qkbase = ((size_t)b * T_ + t0) * H_ + n;
#pragma unroll
  for (int r = 0; r < 8; ++r) {
    int t = r + g * 8;
    Qg[qkbase + (size_t)t * H_] = (_Float16)cq[r];
    Kg[qkbase + (size_t)t * H_] = (_Float16)ck[r];
  }
  // V transposed: lane's 8 rows are contiguous t -> one packed 16B store
  v8h vp;
#pragma unroll
  for (int r = 0; r < 8; ++r) vp[r] = (_Float16)cv[r];
  *(v8h*)(&Vt[((size_t)b * H_ + n) * T_ + t0 + g * 8]) = vp;
}

// ---------------------------------------------------------------------------
// Phase 2: flash attention, one wave per 16-query tile (4 waves/block).
// Sᵀ = K·Qᵀ so P lands directly in the A-fragment layout for P·V.
// Online softmax fully in registers; causal trip count per tile.
// ---------------------------------------------------------------------------
__global__ __launch_bounds__(128)
void flash_attn_kernel(const _Float16* __restrict__ Qg,   // [B][T][H]
                       const _Float16* __restrict__ Kg,   // [B][T][H]
                       const _Float16* __restrict__ Vt,   // [B][H][T]
                       float* __restrict__ out)           // [B][T][H] f32
{
  const int tid  = threadIdx.x;
  const int lane = tid & 31;
  const int g    = lane >> 4;
  const int ln   = lane & 15;
  const int wid  = blockIdx.x * 4 + (tid >> 5);
  const int b    = wid >> 4;
  const int qi   = wid & 15;
  const int q0   = qi << 4;

  // B-fragments of Q (constant across kv loop): lane=query, K=h contiguous
  const _Float16* qrow = Qg + ((size_t)b * T_ + q0 + ln) * H_;
  const v16h bq0 = *(const v16h*)(qrow + 16 * g);        // h 0..31
  const v16h bq1 = *(const v16h*)(qrow + 32 + 16 * g);   // h 32..63

  v8f o0 = {}, o1 = {}, o2 = {}, o3 = {};
  float m_i = -1e30f, l_i = 0.f;
  const float scale  = 0.125f;          // 1/sqrt(64)
  const int   q_lane = q0 + ln;         // query whose stats this lane owns
  const int   rowoff = g * 8;           // this lane's C-layout row base

  const _Float16* kbase = Kg + (size_t)b * T_ * H_;
  const _Float16* vbase = Vt + (size_t)b * H_ * T_;
  const int niter = ((q0 + 15) >> 5) + 1;   // causal: keys 0..q0+15

  for (int it = 0; it < niter; ++it) {
    const int j0 = it * 32;

    // A-fragments of K (two 16-key subtiles x two 32-wide h chunks)
    AFrag a00, a01, a10, a11;
    const _Float16* krow0 = kbase + (size_t)(j0 + ln) * H_;
    const _Float16* krow1 = krow0 + 16 * H_;
    a00.h[0] = *(const v8h*)(krow0 + 16 * g);      a00.h[1] = *(const v8h*)(krow0 + 8 + 16 * g);
    a01.h[0] = *(const v8h*)(krow0 + 32 + 16 * g); a01.h[1] = *(const v8h*)(krow0 + 40 + 16 * g);
    a10.h[0] = *(const v8h*)(krow1 + 16 * g);      a10.h[1] = *(const v8h*)(krow1 + 8 + 16 * g);
    a11.h[0] = *(const v8h*)(krow1 + 32 + 16 * g); a11.h[1] = *(const v8h*)(krow1 + 40 + 16 * g);

    v8f s0 = {}, s1 = {};
    s0 = __builtin_amdgcn_wmma_f32_16x16x32_f16(false, a00.v, false, bq0, (short)0, s0, false, false);
    s0 = __builtin_amdgcn_wmma_f32_16x16x32_f16(false, a01.v, false, bq1, (short)0, s0, false, false);
    s1 = __builtin_amdgcn_wmma_f32_16x16x32_f16(false, a10.v, false, bq0, (short)0, s1, false, false);
    s1 = __builtin_amdgcn_wmma_f32_16x16x32_f16(false, a11.v, false, bq1, (short)0, s1, false, false);

    // scale + causal mask; Sᵀ C-layout: lane col = query, VGPR r = key j0+8g+r
    float sv0[8], sv1[8];
    float smax = -1e30f;
#pragma unroll
    for (int r = 0; r < 8; ++r) {
      const int k0 = j0 + rowoff + r;
      float v0 = (k0      <= q_lane) ? s0[r] * scale : -1e30f;
      float v1 = (k0 + 16 <= q_lane) ? s1[r] * scale : -1e30f;
      sv0[r] = v0; sv1[r] = v1;
      smax = fmaxf(smax, fmaxf(v0, v1));
    }
    smax = fmaxf(smax, __shfl_xor(smax, 16, 32));   // other half of this query's keys
    const float newm  = fmaxf(m_i, smax);
    const float alpha = __expf(m_i - newm);
    m_i = newm;

    float rsum = 0.f;
    v8f p0 = {}, p1 = {};
#pragma unroll
    for (int r = 0; r < 8; ++r) {
      float e0 = __expf(sv0[r] - newm);
      float e1 = __expf(sv1[r] - newm);
      p0[r] = e0; p1[r] = e1;
      rsum += e0 + e1;
    }
    rsum += __shfl_xor(rsum, 16, 32);
    l_i = l_i * alpha + rsum;

    // Pack P(16x32) into the A-fragment: Sᵀ C-layout == A-layout per lane.
    AFrag ap;
#pragma unroll
    for (int i = 0; i < 8; ++i) {
      ap.v[i]     = (_Float16)p0[i];   // K = 8g+i
      ap.v[8 + i] = (_Float16)p1[i];   // K = 16+8g+i
    }

    // Rescale O rows by alpha of their query (broadcast from stat-owner lanes)
#pragma unroll
    for (int r = 0; r < 8; ++r) {
      float ar = __shfl(alpha, rowoff + r, 32);
      o0[r] *= ar; o1[r] *= ar; o2[r] *= ar; o3[r] *= ar;
    }

    // O += P·V ; B-fragments of V from transposed layout: K = 8g+i and 16+8g+i per lane
    const _Float16* vcol = vbase + (size_t)ln * T_ + j0 + 8 * g;
    v16h bv0 = __builtin_shufflevector(*(const v8h*)(vcol), *(const v8h*)(vcol + 16), 0, 1, 2, 3, 4, 5, 6, 7, 8, 9, 10, 11, 12, 13, 14, 15);
    v16h bv1 = __builtin_shufflevector(*(const v8h*)(vcol + 16 * T_), *(const v8h*)(vcol + 16 * T_ + 16), 0, 1, 2, 3, 4, 5, 6, 7, 8, 9, 10, 11, 12, 13, 14, 15);
    v16h bv2 = __builtin_shufflevector(*(const v8h*)(vcol + 32 * T_), *(const v8h*)(vcol + 32 * T_ + 16), 0, 1, 2, 3, 4, 5, 6, 7, 8, 9, 10, 11, 12, 13, 14, 15);
    v16h bv3 = __builtin_shufflevector(*(const v8h*)(vcol + 48 * T_), *(const v8h*)(vcol + 48 * T_ + 16), 0, 1, 2, 3, 4, 5, 6, 7, 8, 9, 10, 11, 12, 13, 14, 15);
    o0 = __builtin_amdgcn_wmma_f32_16x16x32_f16(false, ap.v, false, bv0, (short)0, o0, false, false);
    o1 = __builtin_amdgcn_wmma_f32_16x16x32_f16(false, ap.v, false, bv1, (short)0, o1, false, false);
    o2 = __builtin_amdgcn_wmma_f32_16x16x32_f16(false, ap.v, false, bv2, (short)0, o2, false, false);
    o3 = __builtin_amdgcn_wmma_f32_16x16x32_f16(false, ap.v, false, bv3, (short)0, o3, false, false);
  }

  // Normalize and store fp32 output
  float* orow = out + ((size_t)b * T_ + q0) * H_;
#pragma unroll
  for (int r = 0; r < 8; ++r) {
    float lr  = __shfl(l_i, rowoff + r, 32);
    float inv = 1.0f / lr;
    float* p  = orow + (size_t)(rowoff + r) * H_ + ln;
    p[0]  = o0[r] * inv;
    p[16] = o1[r] * inv;
    p[32] = o2[r] * inv;
    p[48] = o3[r] * inv;
  }
}

// ---------------------------------------------------------------------------
extern "C" void kernel_launch(void* const* d_in, const int* in_sizes, int n_in,
                              void* d_out, int out_size, void* d_ws, size_t ws_size,
                              hipStream_t stream) {
  (void)in_sizes; (void)n_in; (void)out_size; (void)ws_size;
  const float* x  = (const float*)d_in[0];
  const float* Wq = (const float*)d_in[1];
  const float* Wk = (const float*)d_in[2];
  const float* Wv = (const float*)d_in[3];
  float* out = (float*)d_out;

  _Float16* ws = (_Float16*)d_ws;
  _Float16* Wt = ws;                                   // 3*H*C halves
  _Float16* Qg = Wt + 3 * (size_t)H_ * C_;             // B*T*H halves
  _Float16* Kg = Qg + (size_t)B_ * T_ * H_;
  _Float16* Vt = Kg + (size_t)B_ * T_ * H_;            // [B][H][T]

  wt_convert_kernel<<<(3 * C_ * H_ + 255) / 256, 256, 0, stream>>>(Wq, Wk, Wv, Wt);
  qkv_proj_kernel<<<B_ * (T_ / 16), 128, 0, stream>>>(x, Wt, Qg, Kg, Vt);
  flash_attn_kernel<<<B_ * (T_ / 16) / 4, 128, 0, stream>>>(Qg, Kg, Vt, out);
}
